// Model_47493748359815
// MI455X (gfx1250) — matrix operand order repaired and checked by reading, not compiled, not run on hardware
//
// MI455X / gfx1250 implementation.
// Strategy: everything routed through V_WMMA_F32_16X16X32_F16 (f16 A/B, f32 acc).
// ~25 GFLOP fp32-equivalent model; HBM traffic ~70MB (~3us @23.3TB/s) -> compute bound,
// so weights are pre-swizzled once into B-fragment layout (k_prep) and all GEMMs are
// tight WMMA loops with one contiguous 32B B-load per lane per k-step.
#include <hip/hip_runtime.h>

typedef __attribute__((ext_vector_type(16))) _Float16 v16h;
typedef __attribute__((ext_vector_type(8)))  float    v8f;

#define NBATCH 2048
#define HSTEPS 12
#define KROWS  16
#define DFEAT  64

struct Ptrs { const void* p[77]; };

// ---- weight matrix table (prepped into d_ws as f16 B-fragments) -------------
// idx 0..19: branch 64x64 mats (fw/bw interleaved): W12,W2a,W2b,W3a,W3b,W4a,W4b,W5,W6a,W6b
// 20: WTE2  21: Wy1(128x64) 22: Wy2  23: Wx1(128x64) 24: Wx2  25: Wg1 26: Wgl2
// 27: Wg2(ling2)  28: W_ih(128x192)  29: W_hh(64x192)  30: Wo1  31: Wo2
__constant__ int g_matSrc[32] = {7,11, 13,17, 15,19, 21,25, 23,27, 29,33, 31,35,
                                 37,39, 41,45, 43,47, 51, 53,55, 57,59, 61,63,
                                 65, 67,68, 71,73};
__constant__ int g_matK[32]   = {64,64,64,64,64,64,64,64,64,64,64,64,64,64,64,64,
                                 64,64,64,64,64, 128,64, 128,64, 64,64, 64,
                                 128,64, 64,64};
__constant__ int g_matN[32]   = {64,64,64,64,64,64,64,64,64,64,64,64,64,64,64,64,
                                 64,64,64,64,64, 64,64, 64,64, 64,64, 64,
                                 192,192, 64,64};
__constant__ unsigned g_matOff[32] = {
    0,4096,8192,12288,16384,20480,24576,28672,32768,36864,
    40960,45056,49152,53248,57344,61440,65536,69632,73728,77824,
    81920, 86016,94208, 98304,106496, 110592,114688, 118784,
    122880, 147456, 159744, 163840};
#define WTS_HALVES 167936
#define OFF_WTE2 81920u
#define OFF_WY1  86016u
#define OFF_WY2  94208u
#define OFF_WX1  98304u
#define OFF_WX2  106496u
#define OFF_WG1  110592u
#define OFF_WGL2 114688u
#define OFF_WG2  118784u
#define OFF_WIH  122880u
#define OFF_WHH  147456u
#define OFF_WO1  159744u
#define OFF_WO2  163840u

// ---- activation functors ----------------------------------------------------
struct Ident { __device__ float operator()(float v) const { return v; } };
struct Relu  { __device__ float operator()(float v) const { return fmaxf(v, 0.f); } };
struct TanhA { __device__ float operator()(float v) const { return tanhf(v); } };

__device__ __forceinline__ float sigm(float x) { return 1.f / (1.f + __expf(-x)); }

// ---- WMMA core --------------------------------------------------------------
__device__ __forceinline__ v8f wmma16(v16h a, v16h b, v8f c) {
  return __builtin_amdgcn_wmma_f32_16x16x32_f16(false, a, false, b, (short)0, c,
                                                false, false);
}

// A fragment from a row-major f32 LDS tile (16 x KDIM). ISA 16-bit A layout:
// lane half q = lane>>4 : a[i]=A[m][32c+8q+i], a[8+i]=A[m][32c+16+8q+i], m=lane&15.
__device__ __forceinline__ v16h load_a_lds(const float* T, int stride, int kc, int lane) {
  int m  = lane & 15;
  int kb = kc * 32 + ((lane >> 4) << 3);
  const float* p0 = T + m * stride + kb;
  v16h a;
#pragma unroll
  for (int i = 0; i < 8; ++i) {
    a[i]     = (_Float16)p0[i];
    a[i + 8] = (_Float16)p0[16 + i];
  }
  return a;
}

// Row mode: A has only row 0 (a 64-float LDS row); rows 1..15 are zero.
__device__ __forceinline__ v16h load_a_row(const float* row, int kc, int lane) {
  float msk = ((lane & 15) == 0) ? 1.f : 0.f;
  int kb = kc * 32 + ((lane >> 4) << 3);
  v16h a;
#pragma unroll
  for (int i = 0; i < 8; ++i) {
    a[i]     = (_Float16)(msk * row[kb + i]);
    a[i + 8] = (_Float16)(msk * row[kb + 16 + i]);
  }
  return a;
}

// GEMM: (16 x KDIM) LDS tile  @  prepped B (KDIM x 16*NT starting at col tile tOff).
// out[t] = C fragments; init = bias[col] (must be valid when !ACC) or accumulate.
template <int KDIM, int NT, bool ACC>
__device__ __forceinline__ void gemm_core(const float* aT, const _Float16* wts,
                                          size_t off, int tOff, const float* bias,
                                          int lane, v8f* out) {
  constexpr int NC = KDIM / 32;
  v16h af[NC];
#pragma unroll
  for (int kc = 0; kc < NC; ++kc) af[kc] = load_a_lds(aT, KDIM, kc, lane);
#pragma unroll
  for (int t = 0; t < NT; ++t) {
    v8f c;
    if (ACC) {
      c = out[t];
    } else {
      float bv = bias[(lane & 15) + ((t + tOff) << 4)];  // unconditional: no branch
#pragma unroll
      for (int r = 0; r < 8; ++r) c[r] = bv;
    }
#pragma unroll
    for (int kc = 0; kc < NC; ++kc) {
      const v16h* bp = (const v16h*)(wts + off +
                        (size_t)((((t + tOff) * NC) + kc) * 32 + lane) * 16);
      c = wmma16(af[kc], *bp, c);
    }
    out[t] = c;
  }
}

// C frags -> row-major f32 LDS tile (lane=N, VGPR r -> M = 8*(lane>>4)+r).
template <int NT, typename F>
__device__ __forceinline__ void frags_to_lds(const v8f* fr, float* T, int stride,
                                             int lane, F act) {
  int mb = (lane >> 4) << 3;
  int n0 = lane & 15;
#pragma unroll
  for (int t = 0; t < NT; ++t)
#pragma unroll
    for (int r = 0; r < 8; ++r)
      T[(mb + r) * stride + n0 + 16 * t] = act(fr[t][r]);
}

// (1x64) row @ W(64x64) + b, via WMMA with single active A row.
template <typename F>
__device__ __forceinline__ void row_gemm64(const float* rin, const _Float16* wts,
                                           size_t off, const float* bias,
                                           float* rout, int lane, F act) {
  v16h a0 = load_a_row(rin, 0, lane);
  v16h a1 = load_a_row(rin, 1, lane);
#pragma unroll
  for (int t = 0; t < 4; ++t) {
    int n = (lane & 15) + 16 * t;
    v8f c;
    float bv = bias[n];
#pragma unroll
    for (int r = 0; r < 8; ++r) c[r] = bv;
    c = wmma16(a0, *(const v16h*)(wts + off + (size_t)((t * 2 + 0) * 32 + lane) * 16), c);
    c = wmma16(a1, *(const v16h*)(wts + off + (size_t)((t * 2 + 1) * 32 + lane) * 16), c);
    if (lane < 16) rout[n] = act(c[0]);  // row 0 lives in VGPR0 of lanes 0..15
  }
}

// ---- kernel 0: swizzle fp32 weights into f16 B-fragment layout in d_ws ------
__global__ void __launch_bounds__(256) k_prep(Ptrs P, _Float16* wts) {
  int tid = blockIdx.x * blockDim.x + threadIdx.x;
  if (tid >= WTS_HALVES / 16) return;  // 10496 fragment-lane triples
  int m = 0, t = tid;
  while (t >= (g_matK[m] * g_matN[m]) >> 4) { t -= (g_matK[m] * g_matN[m]) >> 4; ++m; }
  const float* src = (const float*)P.p[g_matSrc[m]];
  int Nd = g_matN[m];
  int numC = g_matK[m] >> 5;
  int lane = t & 31, rest = t >> 5;
  int c = rest % numC, tt = rest / numC;
  int nn = (lane & 15) + 16 * tt;
  int kb = c * 32 + ((lane >> 4) << 3);
  _Float16* dst = wts + (size_t)g_matOff[m] + (size_t)t * 16;
#pragma unroll
  for (int i = 0; i < 16; ++i) dst[i] = (_Float16)src[(size_t)(kb + i + ((i >> 3) << 3)) * Nd + nn];
}

// ---- kernel 1: per-(n,h,branch) branch pipeline -----------------------------
__global__ void __launch_bounds__(32) k_branch(Ptrs P, const _Float16* wts,
                                               float* xcat, float* ycat) {
  __shared__ float bufA[16 * 64];
  __shared__ float bufB[16 * 64];
  __shared__ float rowY[64], rowT[64], rowZ[64];
  __shared__ float xin[16], gpv[16];
  const int lane = threadIdx.x;
  const int w = blockIdx.x;
  const int br = w & 1;
  const int nh = w >> 1;                 // n*H + h
  const int n = nh / HSTEPS;
  const float* xg = (const float*)P.p[0 + br];
  const float* gp = (const float*)P.p[3 + br];
  if (lane < 16) {
    xin[lane] = (xg[nh * KROWS + lane] - 50.f) * 0.05f;
    gpv[lane] = gp[n * KROWS + lane];
  }
  __syncthreads();
  const float* W1  = (const float*)P.p[5 + 4 * br];
  const float* b1  = (const float*)P.p[6 + 4 * br];
  const float* b12 = (const float*)P.p[8 + 4 * br];
  const float* b2a = (const float*)P.p[14 + 4 * br];
  const float* b2b = (const float*)P.p[16 + 4 * br];
  const float* b3a = (const float*)P.p[22 + 4 * br];
  const float* b3b = (const float*)P.p[24 + 4 * br];
  const float* b4a = (const float*)P.p[30 + 4 * br];
  const float* b4b = (const float*)P.p[32 + 4 * br];
  const float* b5  = (const float*)P.p[38 + 2 * br];
  const float* b6a = (const float*)P.p[42 + 4 * br];
  const float* b6b = (const float*)P.p[44 + 4 * br];
  const size_t oW12 = (size_t)(0 + br) * 4096,  oW2a = (size_t)(2 + br) * 4096;
  const size_t oW2b = (size_t)(4 + br) * 4096,  oW3a = (size_t)(6 + br) * 4096;
  const size_t oW3b = (size_t)(8 + br) * 4096,  oW4a = (size_t)(10 + br) * 4096;
  const size_t oW4b = (size_t)(12 + br) * 4096, oW5  = (size_t)(14 + br) * 4096;
  const size_t oW6a = (size_t)(16 + br) * 4096, oW6b = (size_t)(18 + br) * 4096;

  // p1.l1: X0[k][d] = relu(xin[k]*W1[d] + b1[d])
  for (int i = lane; i < 1024; i += 32) {
    int k = i >> 6, d = i & 63;
    bufA[i] = fmaxf(fmaf(xin[k], W1[d], b1[d]), 0.f);
  }
  __syncthreads();
  // X = X0 @ W12 + b12  -> bufB
  { v8f f[4]; gemm_core<64, 4, false>(bufA, wts, oW12, 0, b12, lane, f);
    frags_to_lds<4>(f, bufB, 64, lane, Ident()); }
  __syncthreads();
  // y0 = gp @ X
  { float a0 = 0.f, a1 = 0.f;
#pragma unroll
    for (int k = 0; k < 16; ++k) {
      float g = gpv[k];
      a0 = fmaf(g, bufB[k * 64 + lane], a0);
      a1 = fmaf(g, bufB[k * 64 + 32 + lane], a1);
    }
    rowY[lane] = a0; rowY[lane + 32] = a1; }
  __syncthreads();
  // y = two(p2, y0)
  row_gemm64(rowY, wts, oW2a, b2a, rowT, lane, Relu());  __syncthreads();
  row_gemm64(rowT, wts, oW2b, b2b, rowY, lane, Ident()); __syncthreads();
  // H3 = relu(X @ W3a + b3a) -> bufA ; X3 = H3 @ W3b + b3b (frags only)
  { v8f f[4]; gemm_core<64, 4, false>(bufB, wts, oW3a, 0, b3a, lane, f);
    frags_to_lds<4>(f, bufA, 64, lane, Relu()); }
  __syncthreads();
  v8f x3f[4];
  gemm_core<64, 4, false>(bufA, wts, oW3b, 0, b3b, lane, x3f);
  // z = gp @ |y - X3|  (reduce the two M-halves across lanes l <-> l^16)
  { int mb = (lane >> 4) << 3;
#pragma unroll
    for (int t = 0; t < 4; ++t) {
      int nn = (lane & 15) + 16 * t;
      float yv = rowY[nn];
      float acc = 0.f;
#pragma unroll
      for (int r = 0; r < 8; ++r) acc = fmaf(gpv[mb + r], fabsf(yv - x3f[t][r]), acc);
      acc += __shfl_xor(acc, 16, 32);
      if (lane < 16) rowZ[nn] = acc;
    } }
  __syncthreads();
  // x = lin5(tanh(two(p4, z)))
  row_gemm64(rowZ, wts, oW4a, b4a, rowT, lane, Relu());  __syncthreads();
  row_gemm64(rowT, wts, oW4b, b4b, rowZ, lane, TanhA()); __syncthreads();
  row_gemm64(rowZ, wts, oW5,  b5,  rowT, lane, Ident()); __syncthreads();
  xcat[nh * 128 + br * 64 + lane]      = rowT[lane];
  xcat[nh * 128 + br * 64 + 32 + lane] = rowT[32 + lane];
  // y_out = two(p6, x + y)
  rowZ[lane]      = rowT[lane]      + rowY[lane];
  rowZ[lane + 32] = rowT[lane + 32] + rowY[lane + 32];
  __syncthreads();
  row_gemm64(rowZ, wts, oW6a, b6a, rowY, lane, Relu());  __syncthreads();
  row_gemm64(rowY, wts, oW6b, b6b, rowZ, lane, Ident()); __syncthreads();
  ycat[nh * 128 + br * 64 + lane]      = rowZ[lane];
  ycat[nh * 128 + br * 64 + 32 + lane] = rowZ[32 + lane];
}

// ---- kernel 2: TE embedding (12x64) -----------------------------------------
__global__ void __launch_bounds__(32) k_te(Ptrs P, const _Float16* wts, float* te) {
  __shared__ float bufA[16 * 64];
  const int lane = threadIdx.x;
  const int*  TE = (const int*)P.p[2];
  const float* W = (const float*)P.p[49];
  const float* b = (const float*)P.p[50];
  for (int i = lane; i < 1024; i += 32) {
    int hh = i >> 6, d = i & 63;
    bufA[i] = (hh < HSTEPS) ? fmaxf(W[TE[hh] * 64 + d] + b[d], 0.f) : 0.f;
  }
  __syncthreads();
  v8f f[4];
  gemm_core<64, 4, false>(bufA, wts, OFF_WTE2, 0, (const float*)P.p[52], lane, f);
  int mb = (lane >> 4) << 3, n0 = lane & 15;
#pragma unroll
  for (int t = 0; t < 4; ++t)
#pragma unroll
    for (int r = 0; r < 8; ++r)
      if (mb + r < HSTEPS) te[(mb + r) * 64 + n0 + 16 * t] = f[t][r];
}

// ---- kernel 3: liny/linx/ling + yg = exp(-relu(g))*y ------------------------
__global__ void __launch_bounds__(32) k_combine(Ptrs P, const _Float16* wts,
                                                const float* xcat, const float* ycat,
                                                float* yg) {
  __shared__ float bufA[16 * 128];
  __shared__ float bufB[16 * 64];
  const int lane = threadIdx.x;
  const int m0 = blockIdx.x * 16;  // rows over N*H
  const float* by1 = (const float*)P.p[54]; const float* by2 = (const float*)P.p[56];
  const float* bx1 = (const float*)P.p[58]; const float* bx2 = (const float*)P.p[60];
  const float* bg1 = (const float*)P.p[62]; const float* bg2 = (const float*)P.p[64];
  for (int i = lane; i < 2048; i += 32) bufA[i] = ycat[(size_t)m0 * 128 + i];
  __syncthreads();
  { v8f f[4]; gemm_core<128, 4, false>(bufA, wts, OFF_WY1, 0, by1, lane, f);
    frags_to_lds<4>(f, bufB, 64, lane, Relu()); }
  __syncthreads();
  v8f yfr[4];
  gemm_core<64, 4, false>(bufB, wts, OFF_WY2, 0, by2, lane, yfr);
  __syncthreads();
  for (int i = lane; i < 2048; i += 32) bufA[i] = xcat[(size_t)m0 * 128 + i];
  __syncthreads();
  { v8f f[4]; gemm_core<128, 4, false>(bufA, wts, OFF_WX1, 0, bx1, lane, f);
    frags_to_lds<4>(f, bufB, 64, lane, Relu()); }
  __syncthreads();
  { v8f f[4]; gemm_core<64, 4, false>(bufB, wts, OFF_WX2, 0, bx2, lane, f);
    frags_to_lds<4>(f, bufA, 64, lane, Ident()); }   // x tile
  __syncthreads();
  { v8f f[4]; gemm_core<64, 4, false>(bufA, wts, OFF_WG1, 0, bg1, lane, f);
    frags_to_lds<4>(f, bufB, 64, lane, Relu()); }
  __syncthreads();
  v8f gfr[4];
  gemm_core<64, 4, false>(bufB, wts, OFF_WGL2, 0, bg2, lane, gfr);
  int mb = (lane >> 4) << 3, n0 = lane & 15;
#pragma unroll
  for (int t = 0; t < 4; ++t)
#pragma unroll
    for (int r = 0; r < 8; ++r) {
      float g = __expf(-fmaxf(gfr[t][r], 0.f));
      yg[(size_t)(m0 + mb + r) * 64 + n0 + 16 * t] = g * yfr[t][r];
    }
}

// ---- kernel 4: GRU over H steps + output head -------------------------------
__global__ void __launch_bounds__(32) k_gru(Ptrs P, const _Float16* wts,
                                            const float* yg, const float* te,
                                            float* out) {
  __shared__ float sS[16 * 64];   // state
  __shared__ float sT[16 * 64];   // s2 / hidden / p
  __shared__ float sY[16 * 128];  // y_t
  __shared__ float sTe[HSTEPS * 64];
  const int lane = threadIdx.x;
  const int n0 = blockIdx.x * 16;
  const float* bgl = (const float*)P.p[66];
  const float* bih = (const float*)P.p[69];
  const float* bhh = (const float*)P.p[70];
  const float* bo1 = (const float*)P.p[72];
  const float* bo2 = (const float*)P.p[74];
  const float* Wo  = (const float*)P.p[75];
  const float* bo  = (const float*)P.p[76];
  for (int i = lane; i < HSTEPS * 64; i += 32) sTe[i] = te[i];
  for (int i = lane; i < 1024; i += 32) sS[i] = 0.f;
  __syncthreads();
  const int mb = (lane >> 4) << 3, nl = lane & 15;
  for (int h = 0; h < HSTEPS; ++h) {
    // stage y_t = [yg(:,h,:) | te(h,:)]
    for (int i = lane; i < 2048; i += 32) {
      int m = i >> 7, d = i & 127;
      sY[i] = (d < 64) ? yg[((size_t)(n0 + m) * HSTEPS + h) * 64 + d]
                       : sTe[h * 64 + (d - 64)];
    }
    // g2 = exp(-relu(s @ Wg2 + bg2));  s2 = g2 * s
    v8f gf[4];
    gemm_core<64, 4, false>(sS, wts, OFF_WG2, 0, bgl, lane, gf);
    __syncthreads();
#pragma unroll
    for (int t = 0; t < 4; ++t)
#pragma unroll
      for (int r = 0; r < 8; ++r) {
        float g = __expf(-fmaxf(gf[t][r], 0.f));
        sT[(mb + r) * 64 + nl + 16 * t] = g * sS[(mb + r) * 64 + nl + 16 * t];
      }
    __syncthreads();
    // gates: r/z tiles get gx+gh fused via accumulating WMMA; n-gate kept split
    v8f gxz[8], gx2[4], gh2[4];
    gemm_core<128, 8, false>(sY, wts, OFF_WIH, 0, bih, lane, gxz);
#pragma unroll
    for (int t = 0; t < 8; ++t) {
      float bv = bhh[nl + 16 * t];
#pragma unroll
      for (int r = 0; r < 8; ++r) gxz[t][r] += bv;
    }
    gemm_core<64, 8, true>(sT, wts, OFF_WHH, 0, bhh /*unused*/, lane, gxz);
    gemm_core<128, 4, false>(sY, wts, OFF_WIH, 8, bih, lane, gx2);
    gemm_core<64, 4, false>(sT, wts, OFF_WHH, 8, bhh, lane, gh2);
    __syncthreads();
#pragma unroll
    for (int t = 0; t < 4; ++t)
#pragma unroll
      for (int r = 0; r < 8; ++r) {
        float rr = sigm(gxz[t][r]);
        float zz = sigm(gxz[4 + t][r]);
        float nv = tanhf(gx2[t][r] + rr * gh2[t][r]);
        float s2 = sT[(mb + r) * 64 + nl + 16 * t];
        sS[(mb + r) * 64 + nl + 16 * t] = (1.f - zz) * nv + zz * s2;
      }
    __syncthreads();
    // output head: p = relu(relu(s@Wo1+bo1)@Wo2+bo2); pred = p@Wout + bout
    { v8f f[4]; gemm_core<64, 4, false>(sS, wts, OFF_WO1, 0, bo1, lane, f);
      frags_to_lds<4>(f, sT, 64, lane, Relu()); }
    __syncthreads();
    v8f pf[4];
    gemm_core<64, 4, false>(sT, wts, OFF_WO2, 0, bo2, lane, pf);
    __syncthreads();
    frags_to_lds<4>(pf, sT, 64, lane, Relu());
    __syncthreads();
    { int row = lane >> 1, off = (lane & 1) * 32;
      float acc = 0.f;
#pragma unroll
      for (int d = 0; d < 32; ++d) acc = fmaf(sT[row * 64 + off + d], Wo[off + d], acc);
      acc += __shfl_xor(acc, 1, 32);
      if ((lane & 1) == 0)
        out[(size_t)(n0 + row) * HSTEPS + h] = (acc + bo[0]) * 20.f + 50.f; }
    __syncthreads();
  }
}

// ---- host launcher ----------------------------------------------------------
extern "C" void kernel_launch(void* const* d_in, const int* in_sizes, int n_in,
                              void* d_out, int out_size, void* d_ws, size_t ws_size,
                              hipStream_t stream) {
  (void)in_sizes; (void)out_size; (void)ws_size;
  Ptrs P;
  for (int i = 0; i < 77; ++i) P.p[i] = (i < n_in) ? d_in[i] : nullptr;
  char* ws = (char*)d_ws;
  _Float16* wts = (_Float16*)ws;                                   // 335,872 B
  float* te   = (float*)(ws + 335872);                             //   3,072 B
  float* xcat = (float*)(ws + 339968);                             // 12.58 MB
  float* ycat = (float*)(ws + 339968 + 12582912);                  // 12.58 MB
  float* yg   = (float*)(ws + 339968 + 2 * 12582912);              //  6.29 MB
  float* out  = (float*)d_out;

  k_prep   <<<(WTS_HALVES / 16 + 255) / 256, 256, 0, stream>>>(P, wts);
  k_branch <<<NBATCH * HSTEPS * 2, 32, 0, stream>>>(P, wts, xcat, ycat);
  k_te     <<<1, 32, 0, stream>>>(P, wts, te);
  k_combine<<<(NBATCH * HSTEPS) / 16, 32, 0, stream>>>(P, wts, xcat, ycat, yg);
  k_gru    <<<NBATCH / 16, 32, 0, stream>>>(P, wts, yg, te, out);
}
